// TransformerBlock_62414464746025
// MI455X (gfx1250) — matrix operand order repaired and checked by reading, not compiled, not run on hardware
//
#include <hip/hip_runtime.h>
#include <stdint.h>

// ---------------------------------------------------------------------------
// CDNA5 / gfx1250 WMMA types
// ---------------------------------------------------------------------------
typedef __attribute__((ext_vector_type(16))) __bf16          v16bf;
typedef __attribute__((ext_vector_type(16))) unsigned short  v16u;
typedef __attribute__((ext_vector_type(8)))  float           v8f;

static __device__ __forceinline__ unsigned short f2bf(float f) {
  union { float f; unsigned int u; } v; v.f = f;
  unsigned int r = v.u + 0x7FFFu + ((v.u >> 16) & 1u);   // round-to-nearest-even
  return (unsigned short)(r >> 16);
}
static __device__ __forceinline__ float bf2f(unsigned short h) {
  union { unsigned int u; float f; } v; v.u = ((unsigned int)h) << 16;
  return v.f;
}

// ---------------------------------------------------------------------------
// Generic bf16 WMMA GEMM:  D[z] = A[z](MxK) @ B[z](KxN) (+bias)(+*mul)(+add)
//   block = 256 threads = 8 waves. Block tile: 64M x 128N.
//   Wave w: n-subtile w (16 wide) x 4 m-subtiles -> 4 accumulators,
//   4 v_wmma per k-step sharing one B fragment. K multiple of 32.
// ---------------------------------------------------------------------------
template<bool HB, bool HM, bool HA, bool OF, bool OH>
__launch_bounds__(256)
__global__ void gemm_bf16(const unsigned short* __restrict__ A, long long strideA,
                          const unsigned short* __restrict__ Bm, long long strideB,
                          const float* __restrict__ bias,
                          const float* __restrict__ mulp,
                          const float* __restrict__ addp,
                          long long strideC,
                          float* __restrict__ outF,
                          unsigned short* __restrict__ outH,
                          int N, int K)
{
  __shared__ unsigned short sA[64][40];    // [m][k]  row = 80B (b128-aligned)
  __shared__ unsigned short sB[128][36];   // [n][k]  transposed

  const int tid  = threadIdx.x;
  const int lane = tid & 31, wave = tid >> 5;
  const int half = lane >> 4, l16 = lane & 15;
  const int n0 = blockIdx.x * 128;
  const int m0 = blockIdx.y * 64;
  const int z  = blockIdx.z;

  const unsigned short* Ab = A  + (long long)z * strideA;
  const unsigned short* Bb = Bm + (long long)z * strideB;

  const int sa_m  = tid >> 2,   sa_kc = (tid & 3) * 8;       // A: 1 uint4/thread
  const int sb_kq = (tid >> 5) * 4, sb_nc = (tid & 31) * 4;  // B: 4k x 4n/thread

  v8f acc[4] = {};
  for (int kk = 0; kk < K; kk += 32) {
    // ---- stage A tile 64x32: one b128 load + one b128 LDS store ----
    {
      uint4 av = *(const uint4*)(Ab + (long long)(m0 + sa_m) * K + kk + sa_kc);
      *(uint4*)&sA[sa_m][sa_kc] = av;
    }
    // ---- stage B tile 32x128: uint2 loads, register transpose, b64 stores ----
    {
      const unsigned short* src = Bb + (long long)(kk + sb_kq) * N + n0 + sb_nc;
      __builtin_prefetch(src + 32LL * N, 0, 0);    // next k-step (global_prefetch)
      unsigned short vv[4][4];
      #pragma unroll
      for (int kr = 0; kr < 4; ++kr)
        *(uint2*)&vv[kr][0] = *(const uint2*)(src + (long long)kr * N);
      #pragma unroll
      for (int ni = 0; ni < 4; ++ni) {
        unsigned short tp[4] = { vv[0][ni], vv[1][ni], vv[2][ni], vv[3][ni] };
        *(uint2*)&sB[sb_nc + ni][sb_kq] = *(uint2*)tp;
      }
    }
    __syncthreads();

    // ---- B fragment once (contiguous k order used here: N=lane&15, K=2v+16*half) ----
    v16u bu;
    const int nn = wave * 16 + l16;
    #pragma unroll
    for (int p = 0; p < 8; ++p) {
      unsigned int t = *(const unsigned int*)&sB[nn][2 * p + 16 * half];
      bu[2 * p] = (unsigned short)t; bu[2 * p + 1] = (unsigned short)(t >> 16);
    }
    const v16bf bfr = __builtin_bit_cast(v16bf, bu);

    // ---- 4 m-subtiles: A fragment + WMMA each ----
    #pragma unroll
    for (int mt = 0; mt < 4; ++mt) {
      v16u au;
      #pragma unroll
      for (int p = 0; p < 8; ++p) {
        int kb = 2 * p + 16 * half;
        unsigned int t = *(const unsigned int*)&sA[mt * 16 + l16][kb];
        au[2 * p] = (unsigned short)t; au[2 * p + 1] = (unsigned short)(t >> 16);
      }
      acc[mt] = __builtin_amdgcn_wmma_f32_16x16x32_bf16(
          false, __builtin_bit_cast(v16bf, au), false, bfr,
          (short)0, acc[mt], false, false);
    }
    __syncthreads();
  }

  // ---- epilogue (compile-time selected): D: M=r+8*half, N=lane&15 ----
  const int gn = n0 + wave * 16 + l16;
  #pragma unroll
  for (int mt = 0; mt < 4; ++mt) {
    #pragma unroll
    for (int r = 0; r < 8; ++r) {
      int gm = m0 + mt * 16 + r + 8 * half;
      float v = acc[mt][r];
      if constexpr (HB) v += bias[gm];
      long long idx = (long long)z * strideC + (long long)gm * N + gn;
      if constexpr (HM) v *= mulp[idx];
      if constexpr (HA) v += addp[idx];
      if constexpr (OF) outF[idx] = v;
      if constexpr (OH) outH[idx] = f2bf(v);
    }
  }
}

// ---------------------------------------------------------------------------
// Dense 3x3 stride-2 reflect conv as implicit GEMM (k = conv(xn, k_w)).
//   M=Cout=128 (64/block), K = 9 taps x 128 ci (tap-major), N = 128x128/batch.
//   Block: one output row (ho), 64 out-channels, 128 wo; 4 WMMAs per k-step.
//   pad=1, stride=2, W=256: only index -1 reflects (-> 1); max index is 255.
// ---------------------------------------------------------------------------
__launch_bounds__(256)
__global__ void conv3x3s2_gemm(const unsigned short* __restrict__ W,   // [co][ci*9+tap] bf16
                               const unsigned short* __restrict__ X,   // (B,128,256,256) bf16
                               float* __restrict__ Out)                // (B,128,128,128) f32
{
  __shared__ unsigned short sA[64][40];
  __shared__ unsigned short sB[128][36];

  const int tid  = threadIdx.x;
  const int lane = tid & 31, wave = tid >> 5;
  const int half = lane >> 4, l16 = lane & 15;
  const int ho  = blockIdx.x;          // 0..127
  const int co0 = blockIdx.y * 64;     // 2 tiles of out channels
  const int b   = blockIdx.z;
  const unsigned short* Xb = X + (long long)b * 128 * 65536;

  const int sa_m  = tid >> 2,   sa_kc = (tid & 3) * 8;       // A: 8 gathers/thread
  const int sb_kp = (tid >> 4) * 2, sb_wo = (tid & 15) * 8;  // B: 2k x 8wo/thread

  v8f acc[4] = {};
  for (int tap = 0; tap < 9; ++tap) {
    const int ty = tap / 3, tx = tap % 3;
    int y = 2 * ho - 1 + ty; if (y < 0) y = 1;       // reflect
    for (int c0 = 0; c0 < 128; c0 += 32) {
      { // stage A: gather 8 weights (stride 9), one b128 LDS store
        long long base = (long long)(co0 + sa_m) * 1152 + tap;
        unsigned short tp[8];
        #pragma unroll
        for (int j = 0; j < 8; ++j)
          tp[j] = W[base + (long long)(c0 + sa_kc + j) * 9];
        *(uint4*)&sA[sa_m][sa_kc] = *(uint4*)tp;
      }
      { // stage B (im2col: k=ci, n=wo; stride-2 gather; paired b32 stores)
        const unsigned short* r0 =
            Xb + (long long)(c0 + sb_kp) * 65536 + (long long)y * 256;
        const unsigned short* r1 = r0 + 65536;
        #pragma unroll
        for (int i = 0; i < 8; ++i) {
          int wo = sb_wo + i;
          int x = 2 * wo - 1 + tx; if (x < 0) x = 1;
          unsigned short tp[2] = { r0[x], r1[x] };
          *(unsigned int*)&sB[wo][sb_kp] = *(unsigned int*)tp;
        }
      }
      __syncthreads();

      v16u bu;
      const int nn = wave * 16 + l16;
      #pragma unroll
      for (int p = 0; p < 8; ++p) {
        unsigned int t = *(const unsigned int*)&sB[nn][2 * p + 16 * half];
        bu[2 * p] = (unsigned short)t; bu[2 * p + 1] = (unsigned short)(t >> 16);
      }
      const v16bf bfr = __builtin_bit_cast(v16bf, bu);

      #pragma unroll
      for (int mt = 0; mt < 4; ++mt) {
        v16u au;
        #pragma unroll
        for (int p = 0; p < 8; ++p) {
          int kb = 2 * p + 16 * half;
          unsigned int t = *(const unsigned int*)&sA[mt * 16 + l16][kb];
          au[2 * p] = (unsigned short)t; au[2 * p + 1] = (unsigned short)(t >> 16);
        }
        acc[mt] = __builtin_amdgcn_wmma_f32_16x16x32_bf16(
            false, __builtin_bit_cast(v16bf, au), false, bfr,
            (short)0, acc[mt], false, false);
      }
      __syncthreads();
    }
  }
  const int wo = wave * 16 + l16;
  #pragma unroll
  for (int mt = 0; mt < 4; ++mt) {
    #pragma unroll
    for (int r = 0; r < 8; ++r) {
      int co = co0 + mt * 16 + r + 8 * half;
      Out[(((long long)b * 128 + co) * 128 + ho) * 128 + wo] = acc[mt][r];
    }
  }
}

// ---------------------------------------------------------------------------
// Channel LayerNorm (over C=128) on (B,128,256,256) f32 -> bf16
// ---------------------------------------------------------------------------
__global__ void layernorm_bf16(const float* __restrict__ X,
                               const float* __restrict__ w,
                               const float* __restrict__ bs,
                               unsigned short* __restrict__ Out)
{
  long long idx = (long long)blockIdx.x * blockDim.x + threadIdx.x; // B*65536
  int p = (int)(idx & 65535);
  int b = (int)(idx >> 16);
  const float* xb = X + (long long)b * 128 * 65536 + p;
  float s = 0.f;
  for (int c = 0; c < 128; ++c) s += xb[(long long)c * 65536];
  float mu = s * (1.0f / 128.0f);
  float vs = 0.f;
  for (int c = 0; c < 128; ++c) { float d = xb[(long long)c * 65536] - mu; vs += d * d; }
  float rstd = rsqrtf(vs * (1.0f / 128.0f) + 1e-5f);
  unsigned short* ob = Out + (long long)b * 128 * 65536 + p;
  for (int c = 0; c < 128; ++c) {
    float v = (xb[(long long)c * 65536] - mu) * rstd * w[c] + bs[c];
    ob[(long long)c * 65536] = f2bf(v);
  }
}

// ---------------------------------------------------------------------------
// Depthwise 3x3 stride-2 reflect: computes q (f32) and a1 (bf16) in one pass
// ---------------------------------------------------------------------------
__global__ void dw3x3s2_qa1(const unsigned short* __restrict__ Xbf,
                            const float* __restrict__ qw,
                            const float* __restrict__ aw,
                            float* __restrict__ Q,
                            unsigned short* __restrict__ A1)
{
  long long idx = (long long)blockIdx.x * blockDim.x + threadIdx.x; // B*128*128*128
  int wo = (int)(idx & 127);
  int ho = (int)((idx >> 7) & 127);
  int c  = (int)((idx >> 14) & 127);
  int b  = (int)(idx >> 21);
  const unsigned short* xc = Xbf + ((long long)(b * 128 + c)) * 65536;
  float aq = 0.f, aa = 0.f;
  #pragma unroll
  for (int ty = 0; ty < 3; ++ty) {
    int y = 2 * ho - 1 + ty; if (y < 0) y = 1;
    #pragma unroll
    for (int tx = 0; tx < 3; ++tx) {
      int x = 2 * wo - 1 + tx; if (x < 0) x = 1;
      float v = bf2f(xc[y * 256 + x]);
      aq += v * qw[c * 9 + ty * 3 + tx];
      aa += v * aw[c * 9 + ty * 3 + tx];
    }
  }
  Q[idx]  = aq;
  A1[idx] = f2bf(aa);
}

// ---------------------------------------------------------------------------
// FFN depthwise 3x3 pad=1 (zeros) + bias + ReLU on (B,256,256,256)
// ---------------------------------------------------------------------------
__global__ void dw3x3_relu(const unsigned short* __restrict__ Y1,
                           const float* __restrict__ w,
                           const float* __restrict__ bs,
                           unsigned short* __restrict__ Y2)
{
  long long idx = (long long)blockIdx.x * blockDim.x + threadIdx.x; // B*256*65536
  int x0 = (int)(idx & 255);
  int y0 = (int)((idx >> 8) & 255);
  int c  = (int)((idx >> 16) & 255);
  int b  = (int)(idx >> 24);
  const unsigned short* yc = Y1 + ((long long)(b * 256 + c)) * 65536;
  float a = bs[c];
  #pragma unroll
  for (int ty = 0; ty < 3; ++ty) {
    int y = y0 - 1 + ty;
    if (y < 0 || y > 255) continue;
    #pragma unroll
    for (int tx = 0; tx < 3; ++tx) {
      int x = x0 - 1 + tx;
      if (x < 0 || x > 255) continue;
      a += bf2f(yc[y * 256 + x]) * w[c * 9 + ty * 3 + tx];
    }
  }
  Y2[idx] = f2bf(a > 0.f ? a : 0.f);
}

// ---------------------------------------------------------------------------
// Small attention math per (b,head): l2-normed gram matrices over l4=16384,
// softmaxes, then P = attn_a(16x8) @ attn_k(8x16), stored 16x16 f32.
// ---------------------------------------------------------------------------
#define ATL 256
__launch_bounds__(256)
__global__ void attn_small(const float* __restrict__ Q,
                           const float* __restrict__ Kt,
                           const float* __restrict__ A,
                           const float* __restrict__ ta,
                           const float* __restrict__ tv,
                           float* __restrict__ Pout)
{
  __shared__ float sq[16][ATL];
  __shared__ float sk[16][ATL];
  __shared__ float sa[8][ATL];
  __shared__ float sQA[16][8], sAK[8][16];
  __shared__ float sQN[16], sKN[16], sAN[8];
  __shared__ float sAttnA[16][8], sAttnK[8][16];

  const int tid = threadIdx.x;
  const int bh  = blockIdx.x;            // b*8 + h
  const int b = bh >> 3, h = bh & 7;
  const float* qb = Q  + ((long long)b * 128 + h * 16) * 16384;
  const float* kb = Kt + ((long long)b * 128 + h * 16) * 16384;
  const float* ab = A  + ((long long)b * 64  + h * 8 ) * 16384;

  float accD = 0.f;   // pairwise dot accumulator
  float accN = 0.f;   // norm accumulator (threads 0..39)

  for (int l0 = 0; l0 < 16384; l0 += ATL) {
    for (int i = tid; i < 16 * ATL; i += 256) {
      int r = i >> 8, cdx = i & 255;
      sq[r][cdx] = qb[(long long)r * 16384 + l0 + cdx];
      sk[r][cdx] = kb[(long long)r * 16384 + l0 + cdx];
    }
    for (int i = tid; i < 8 * ATL; i += 256) {
      sa[i >> 8][i & 255] = ab[(long long)(i >> 8) * 16384 + l0 + (i & 255)];
    }
    __syncthreads();
    if (tid < 128) {                     // q . a  (16x8)
      int c = tid >> 3, d = tid & 7;
      float s = 0.f;
      for (int l = 0; l < ATL; ++l) s += sq[c][l] * sa[d][l];
      accD += s;
    } else {                             // a . k  (8x16)
      int t = tid - 128; int d = t >> 4, c = t & 15;
      float s = 0.f;
      for (int l = 0; l < ATL; ++l) s += sa[d][l] * sk[c][l];
      accD += s;
    }
    if (tid < 16)      { float s = 0.f; for (int l = 0; l < ATL; ++l) s += sq[tid][l] * sq[tid][l]; accN += s; }
    else if (tid < 32) { int r = tid - 16; float s = 0.f; for (int l = 0; l < ATL; ++l) s += sk[r][l] * sk[r][l]; accN += s; }
    else if (tid < 40) { int r = tid - 32; float s = 0.f; for (int l = 0; l < ATL; ++l) s += sa[r][l] * sa[r][l]; accN += s; }
    __syncthreads();
  }

  if (tid < 128) sQA[tid >> 3][tid & 7] = accD;
  else { int t = tid - 128; sAK[t >> 4][t & 15] = accD; }
  if (tid < 16)      sQN[tid]      = accN;
  else if (tid < 32) sKN[tid - 16] = accN;
  else if (tid < 40) sAN[tid - 32] = accN;
  __syncthreads();

  if (tid < 16) {                        // softmax rows of attn_a (over d=8)
    int c = tid;
    float qn = fmaxf(sqrtf(sQN[c]), 1e-12f);
    float row[8]; float mx = -1e30f;
    for (int d = 0; d < 8; ++d) {
      float an = fmaxf(sqrtf(sAN[d]), 1e-12f);
      row[d] = (sQA[c][d] / (qn * an)) * ta[h];
      mx = fmaxf(mx, row[d]);
    }
    float ss = 0.f;
    for (int d = 0; d < 8; ++d) { row[d] = __expf(row[d] - mx); ss += row[d]; }
    for (int d = 0; d < 8; ++d) sAttnA[c][d] = row[d] / ss;
  } else if (tid < 24) {                 // softmax rows of attn_k (over c=16)
    int d = tid - 16;
    float an = fmaxf(sqrtf(sAN[d]), 1e-12f);
    float row[16]; float mx = -1e30f;
    for (int c = 0; c < 16; ++c) {
      float kn = fmaxf(sqrtf(sKN[c]), 1e-12f);
      row[c] = (sAK[d][c] / (an * kn)) * tv[h];
      mx = fmaxf(mx, row[c]);
    }
    float ss = 0.f;
    for (int c = 0; c < 16; ++c) { row[c] = __expf(row[c] - mx); ss += row[c]; }
    for (int c = 0; c < 16; ++c) sAttnK[d][c] = row[c] / ss;
  }
  __syncthreads();

  int c = tid >> 4, cc = tid & 15;       // P = attn_a @ attn_k  (16x16)
  float p = 0.f;
  for (int d = 0; d < 8; ++d) p += sAttnA[c][d] * sAttnK[d][cc];
  Pout[(long long)bh * 256 + c * 16 + cc] = p;
}

// ---------------------------------------------------------------------------
// E[b] = po_w @ blockdiag(P[b,0..7])   (128x128 bf16 per batch)
// Folds the attention-combine and po 1x1 conv into one GEMM operand.
// ---------------------------------------------------------------------------
__global__ void fuse_po_P(const float* __restrict__ po,   // 128x128 f32
                          const float* __restrict__ P,    // (32,16,16) f32
                          unsigned short* __restrict__ E) // (4,128,128) bf16
{
  int i = blockIdx.x * 256 + threadIdx.x;    // 4*128*128
  int c = i & 127;
  int r = (i >> 7) & 127;
  int b = i >> 14;
  int h = c >> 4, k = c & 15;
  const float* Pb = P + ((long long)(b * 8 + h)) * 256;
  const float* pr = po + (long long)r * 128 + h * 16;
  float s = 0.f;
  #pragma unroll
  for (int j = 0; j < 16; ++j) s += pr[j] * Pb[j * 16 + k];
  E[i] = f2bf(s);
}

// ---------------------------------------------------------------------------
__global__ void f32_to_bf16(const float* __restrict__ in,
                            unsigned short* __restrict__ out, int n)
{
  int i = blockIdx.x * blockDim.x + threadIdx.x;
  if (i < n) out[i] = f2bf(in[i]);
}

// ---------------------------------------------------------------------------
extern "C" void kernel_launch(void* const* d_in, const int* in_sizes, int n_in,
                              void* d_out, int out_size, void* d_ws, size_t ws_size,
                              hipStream_t stream)
{
  const float* x        = (const float*)d_in[0];
  const float* illu     = (const float*)d_in[1];
  const float* ln1_w    = (const float*)d_in[2];
  const float* ln1_b    = (const float*)d_in[3];
  const float* q_w      = (const float*)d_in[4];
  const float* k_w      = (const float*)d_in[5];
  const float* v_w      = (const float*)d_in[6];
  const float* a1_w     = (const float*)d_in[7];
  const float* a2_w     = (const float*)d_in[8];
  const float* a2_b     = (const float*)d_in[9];
  const float* temp_a   = (const float*)d_in[10];
  const float* temp_v   = (const float*)d_in[11];
  const float* po_w     = (const float*)d_in[12];
  const float* ln2_w    = (const float*)d_in[13];
  const float* ln2_b    = (const float*)d_in[14];
  const float* p1_w     = (const float*)d_in[15];
  const float* p1_b     = (const float*)d_in[16];
  const float* dw_w     = (const float*)d_in[17];
  const float* dw_b     = (const float*)d_in[18];
  const float* p2_w     = (const float*)d_in[19];
  const float* p2_b     = (const float*)d_in[20];
  float* outp = (float*)d_out;

  // ---- workspace bump allocator (256B aligned) with region reuse ----
  char* ws = (char*)d_ws;
  size_t o = 0;
  auto alloc = [&](size_t bytes) { size_t r = o; o = (o + bytes + 255) & ~(size_t)255; return r; };

  const long long HW = 65536, HW4 = 16384;
  size_t o_wk  = alloc(147456ULL * 2);   // k_w bf16
  size_t o_wv  = alloc(16384ULL * 2);    // v_w bf16
  size_t o_wa2 = alloc(8192ULL * 2);     // a2_w bf16
  size_t o_wp1 = alloc(32768ULL * 2);    // p1_w bf16
  size_t o_wp2 = alloc(32768ULL * 2);    // p2_w bf16
  size_t o_xn  = alloc(4ULL * 128 * HW * 2);   // xn bf16   (reused as ln2-out)
  size_t o_q   = alloc(4ULL * 128 * HW4 * 4);  // q f32     (region reused for y1)
  size_t o_k   = alloc(4ULL * 128 * HW4 * 4);  // k f32
  size_t o_a1  = alloc(4ULL * 128 * HW4 * 2);  // a1 bf16
  size_t o_a   = alloc(4ULL * 64  * HW4 * 4);  // a f32
  size_t o_vr  = alloc(4ULL * 128 * HW * 2);   // v*illu bf16
  size_t o_P   = alloc(32ULL * 256 * 4);       // P 16x16 f32 per (b,h)
  size_t o_E   = alloc(4ULL * 128 * 128 * 2);  // fused po@P bf16 per batch
  size_t o_x2  = alloc(4ULL * 128 * HW * 4);   // residual stream f32
  size_t o_y2  = alloc(4ULL * 256 * HW * 2);   // ffn dw+relu bf16
  size_t o_yln = o_xn;                         // ln2 output reuses xn region
  size_t o_y1  = o_q;                          // p1 output reuses q..vr region (134MB <= 167MB)
  (void)ws_size; (void)in_sizes; (void)n_in; (void)out_size;

  unsigned short* wk  = (unsigned short*)(ws + o_wk);
  unsigned short* wv  = (unsigned short*)(ws + o_wv);
  unsigned short* wa2 = (unsigned short*)(ws + o_wa2);
  unsigned short* wp1 = (unsigned short*)(ws + o_wp1);
  unsigned short* wp2 = (unsigned short*)(ws + o_wp2);
  unsigned short* xn  = (unsigned short*)(ws + o_xn);
  float*          qf  = (float*)(ws + o_q);
  float*          kf  = (float*)(ws + o_k);
  unsigned short* a1  = (unsigned short*)(ws + o_a1);
  float*          af  = (float*)(ws + o_a);
  unsigned short* vr  = (unsigned short*)(ws + o_vr);
  float*          Pb  = (float*)(ws + o_P);
  unsigned short* Eb  = (unsigned short*)(ws + o_E);
  float*          x2  = (float*)(ws + o_x2);
  unsigned short* yln = (unsigned short*)(ws + o_yln);
  unsigned short* y1  = (unsigned short*)(ws + o_y1);
  unsigned short* y2  = (unsigned short*)(ws + o_y2);

  // ---- 1. weight conversions ----
  f32_to_bf16<<<(147456 + 255) / 256, 256, 0, stream>>>(k_w, wk, 147456);
  f32_to_bf16<<<(16384 + 255) / 256, 256, 0, stream>>>(v_w, wv, 16384);
  f32_to_bf16<<<(8192 + 255) / 256, 256, 0, stream>>>(a2_w, wa2, 8192);
  f32_to_bf16<<<(32768 + 255) / 256, 256, 0, stream>>>(p1_w, wp1, 32768);
  f32_to_bf16<<<(32768 + 255) / 256, 256, 0, stream>>>(p2_w, wp2, 32768);

  // ---- 2. LN1 ----
  layernorm_bf16<<<(4 * 65536) / 256, 256, 0, stream>>>(x, ln1_w, ln1_b, xn);

  // ---- 3. depthwise q & a1 (stride2, reflect) ----
  dw3x3s2_qa1<<<(4 * 128 * 16384) / 256, 256, 0, stream>>>(xn, q_w, a1_w, qf, a1);

  // ---- 4. dense k conv (implicit WMMA GEMM, 64 co / block) ----
  conv3x3s2_gemm<<<dim3(128, 2, 4), 256, 0, stream>>>(wk, xn, kf);

  // ---- 5. a = a2_w @ a1 + a2_b   (M=64,K=128,N=16384) ----
  gemm_bf16<true,false,false,true,false><<<dim3(128, 1, 4), 256, 0, stream>>>(
      wa2, 0LL, a1, 128LL * HW4, a2_b, nullptr, nullptr,
      64LL * HW4, af, nullptr, (int)HW4, 128);

  // ---- 6. v = (v_w @ xn) * illu   (M=128,K=128,N=65536) -> bf16 ----
  gemm_bf16<false,true,false,false,true><<<dim3(512, 2, 4), 256, 0, stream>>>(
      wv, 0LL, xn, 128LL * HW, nullptr, illu, nullptr,
      128LL * HW, nullptr, vr, (int)HW, 128);

  // ---- 7. per-head attention -> P = attn_a @ attn_k (16x16 f32) ----
  attn_small<<<32, 256, 0, stream>>>(qf, kf, af, temp_a, temp_v, Pb);

  // ---- 8. E[b] = po_w @ blockdiag(P[b])  (tiny fold) ----
  fuse_po_P<<<(4 * 128 * 128) / 256, 256, 0, stream>>>(po_w, Pb, Eb);

  // ---- 9. x2 = x + E[b] @ vr   (replaces P@v and po GEMMs) ----
  gemm_bf16<false,false,true,true,false><<<dim3(512, 2, 4), 256, 0, stream>>>(
      Eb, 128LL * 128, vr, 128LL * HW, nullptr, nullptr, x,
      128LL * HW, x2, nullptr, (int)HW, 128);

  // ---- 10. LN2 ----
  layernorm_bf16<<<(4 * 65536) / 256, 256, 0, stream>>>(x2, ln2_w, ln2_b, yln);

  // ---- 11. y1 = p1_w @ ln2 + p1_b   (M=256) -> bf16 ----
  gemm_bf16<true,false,false,false,true><<<dim3(512, 4, 4), 256, 0, stream>>>(
      wp1, 0LL, yln, 128LL * HW, p1_b, nullptr, nullptr,
      256LL * HW, nullptr, y1, (int)HW, 128);

  // ---- 12. depthwise 3x3 + bias + relu ----
  dw3x3_relu<<<(4 * 256 * 65536) / 256, 256, 0, stream>>>(y1, dw_w, dw_b, y2);

  // ---- 13. out = x2 + p2_w @ y2 + p2_b   (K=256) -> f32 d_out ----
  gemm_bf16<true,false,true,true,false><<<dim3(512, 2, 4), 256, 0, stream>>>(
      wp2, 0LL, y2, 256LL * HW, p2_b, nullptr, x2,
      128LL * HW, outp, nullptr, (int)HW, 256);
}
